// RNN_10136122819060
// MI455X (gfx1250) — matrix operand order repaired and checked by reading, not compiled, not run on hardware
//
#include <hip/hip_runtime.h>
#include <hip/hip_bf16.h>

#define L_ 4096
#define D_ 2048
#define H_ 2048
#define MDIM 4096
#define NDIM 2048
#define KDIM 2048
#define BM 128
#define BN 128
#define BK 64
#define SCAN_BLOCKS (H_ / 32)   // 64 persistent workgroups, 32 columns each

typedef __attribute__((ext_vector_type(16))) __bf16 v16bf;
typedef __attribute__((ext_vector_type(8)))  float  v8f;

struct Frag32 { uint4 lo, hi; };   // 32 bytes == v16bf

__device__ __forceinline__ unsigned short f2bf(float f) {
    unsigned u = __builtin_bit_cast(unsigned, f);
    u += 0x7FFFu + ((u >> 16) & 1u);          // round-to-nearest-even
    return (unsigned short)(u >> 16);
}

// ---------------------------------------------------------------------------
// C[4096,2048] = A[4096,2048] * B[2048,2048]; fp32 in/out, bf16 WMMA f32-acc.
// Block tile 128x128, BK=64 (two 16x16x32 k-substeps per LDS stage).
// 256 threads = 8 waves in a 2x4 grid; wave tile 64x32 = 4x2 accumulators.
// 16 WMMAs per barrier pair.
// ---------------------------------------------------------------------------
__global__ __launch_bounds__(256)
void gemm_bf16_wmma(const float* __restrict__ A, const float* __restrict__ B,
                    float* __restrict__ C)
{
    __shared__ __align__(16) unsigned short As[BM * BK];   // [m][k]  16KB
    __shared__ __align__(16) unsigned short Bs[BN * BK];   // [n][k]  16KB (B^T)

    const int tid  = threadIdx.x;
    const int lane = tid & 31;
    const int wave = tid >> 5;
    const int wm   = wave >> 2;      // 0..1
    const int wn   = wave & 3;       // 0..3
    const int l15  = lane & 15;
    const int lhi  = lane >> 4;      // 0 or 1
    const int m0   = blockIdx.y * BM;
    const int n0   = blockIdx.x * BN;

    v8f acc[4][2];
#pragma unroll
    for (int i = 0; i < 4; ++i)
#pragma unroll
        for (int j = 0; j < 2; ++j)
            acc[i][j] = (v8f)0.0f;

    for (int k0 = 0; k0 < KDIM; k0 += BK) {
        // ---- stage A tile (128x64 f32 -> bf16), coalesced float4 loads
#pragma unroll
        for (int t = 0; t < 8; ++t) {
            int idx = tid + t * 256;           // 0..2047 float4s
            int row = idx >> 4;                // 16 float4 per row
            int c4  = idx & 15;
            float4 v = *(const float4*)(A + (size_t)(m0 + row) * KDIM + k0 + c4 * 4);
            int o = row * BK + c4 * 4;
            As[o + 0] = f2bf(v.x); As[o + 1] = f2bf(v.y);
            As[o + 2] = f2bf(v.z); As[o + 3] = f2bf(v.w);
        }
        // ---- stage B tile (64x128 f32) transposed into LDS as bf16 [n][k]
#pragma unroll
        for (int t = 0; t < 8; ++t) {
            int idx = tid + t * 256;           // 0..2047 float4s
            int kr  = idx >> 5;                // 32 float4 per k-row
            int c4  = idx & 31;
            float4 v = *(const float4*)(B + (size_t)(k0 + kr) * NDIM + n0 + c4 * 4);
            int nb = c4 * 4;
            Bs[(nb + 0) * BK + kr] = f2bf(v.x);
            Bs[(nb + 1) * BK + kr] = f2bf(v.y);
            Bs[(nb + 2) * BK + kr] = f2bf(v.z);
            Bs[(nb + 3) * BK + kr] = f2bf(v.w);
        }
        __syncthreads();

#pragma unroll
        for (int kk = 0; kk < BK; kk += 32) {
            // fragment loads (ds_load_b128 pairs)
            v16bf afr[4], bfr[2];
#pragma unroll
            for (int mt = 0; mt < 4; ++mt) {
                int row = wm * 64 + mt * 16 + l15;
                const unsigned short* p = &As[row * BK + kk + lhi * 8];
                Frag32 f;
                f.lo = *(const uint4*)(p);
                f.hi = *(const uint4*)(p + 16);     // K = khalf+16..+23
                afr[mt] = __builtin_bit_cast(v16bf, f);
            }
#pragma unroll
            for (int nt = 0; nt < 2; ++nt) {
                int col = wn * 32 + nt * 16 + l15;
                const unsigned short* p = &Bs[col * BK + kk + lhi * 8];
                Frag32 f;
                f.lo = *(const uint4*)(p);
                f.hi = *(const uint4*)(p + 16);     // K = khalf+16..+23
                bfr[nt] = __builtin_bit_cast(v16bf, f);
            }
#pragma unroll
            for (int mt = 0; mt < 4; ++mt)
#pragma unroll
                for (int nt = 0; nt < 2; ++nt)
                    acc[mt][nt] = __builtin_amdgcn_wmma_f32_16x16x32_bf16(
                        false, afr[mt], false, bfr[nt], (short)0, acc[mt][nt],
                        false, false);
        }
        __syncthreads();
    }

    // ---- epilogue: one base pointer, compile-time row/col offsets so stores
    //      fold into global_store_b32 immediates (no per-store 64-bit muls)
    float* cp = C + (size_t)(m0 + wm * 64 + lhi * 8) * NDIM
                  + (n0 + wn * 32 + l15);
#pragma unroll
    for (int mt = 0; mt < 4; ++mt)
#pragma unroll
        for (int nt = 0; nt < 2; ++nt)
#pragma unroll
            for (int r = 0; r < 8; ++r)
                cp[(size_t)(mt * 16 + r) * NDIM + nt * 16] = acc[mt][nt][r];
}

// ---------------------------------------------------------------------------
// Persistent sequential scan: h = tanh(h @ W + xproj[t]).
// 64 workgroups x 256 threads; WG b owns columns [32b, 32b+32).
// Mapping: col = lane (coalesced 128B W-row reads), k-partition = wave.
// Cross-wave reduction via small LDS tile; device-wide barrier per step.
// ---------------------------------------------------------------------------
__global__ __launch_bounds__(256)
void rnn_scan(const float* __restrict__ xproj, const float* __restrict__ h0,
              const float* __restrict__ wh, float* __restrict__ hs,
              unsigned* __restrict__ bar)
{
    __shared__ float hsh[H_];          // current h (8KB)
    __shared__ float partial[8][32];   // per-wave partials
    const int tid   = threadIdx.x;
    const int lane  = tid & 31;
    const int wv    = tid >> 5;                // 0..7
    const int n0    = blockIdx.x * 32;
    const int kbase = wv * (H_ / 8);           // 256 k's per wave

    for (int step = 0; step < L_; ++step) {
        const float* hsrc = (step == 0) ? h0 : (hs + (size_t)(step - 1) * H_);
        for (int i = tid; i < H_ / 4; i += 256)
            ((float4*)hsh)[i] = ((const float4*)hsrc)[i];
        __syncthreads();

        float acc = 0.0f;
        const float* wp = wh + (size_t)kbase * H_ + n0 + lane;
#pragma unroll 8
        for (int i = 0; i < H_ / 8; ++i)
            acc = fmaf(hsh[kbase + i], wp[(size_t)i * H_], acc);
        partial[wv][lane] = acc;
        __syncthreads();

        if (wv == 0) {
            float s = ((partial[0][lane] + partial[1][lane]) +
                       (partial[2][lane] + partial[3][lane])) +
                      ((partial[4][lane] + partial[5][lane]) +
                       (partial[6][lane] + partial[7][lane]));
            float v = tanhf(s + xproj[(size_t)step * H_ + n0 + lane]);
            hs[(size_t)step * H_ + n0 + lane] = v;
        }
        __threadfence();       // make this step's h visible device-wide
        __syncthreads();

        if (tid == 0) {
            unsigned old = atomicAdd(&bar[0], 1u);
            if (old == (unsigned)(SCAN_BLOCKS - 1)) {
                atomicSub(&bar[0], (unsigned)SCAN_BLOCKS);
                __threadfence();
                atomicAdd(&bar[1], 1u);        // release generation step+1
            } else {
                while (__hip_atomic_load(&bar[1], __ATOMIC_RELAXED,
                                         __HIP_MEMORY_SCOPE_AGENT)
                       < (unsigned)(step + 1)) {
                    __builtin_amdgcn_s_sleep(2);
                }
            }
            __threadfence();   // acquire: invalidate WGP cache for next h read
        }
        __syncthreads();
    }
}

__global__ __launch_bounds__(256)
void copy_hfinal(const float* __restrict__ hs, float* __restrict__ dst)
{
    int i = blockIdx.x * 256 + threadIdx.x;
    if (i < H_) dst[i] = hs[(size_t)(L_ - 1) * H_ + i];
}

// ---------------------------------------------------------------------------
extern "C" void kernel_launch(void* const* d_in, const int* in_sizes, int n_in,
                              void* d_out, int out_size, void* d_ws, size_t ws_size,
                              hipStream_t stream)
{
    (void)in_sizes; (void)n_in; (void)out_size; (void)ws_size;
    const float* x  = (const float*)d_in[0];   // [L, D]
    const float* h0 = (const float*)d_in[1];   // [1, H]
    const float* wh = (const float*)d_in[2];   // [H, H]
    const float* wi = (const float*)d_in[3];   // [D, H]
    const float* wo = (const float*)d_in[4];   // [H, H]
    float* out = (float*)d_out;                // outputs [L,H] then h_final [H]

    float*    xproj = (float*)d_ws;                       // L*H f32 (32 MB)
    float*    hs    = xproj + (size_t)L_ * H_;            // L*H f32 (32 MB)
    unsigned* bar   = (unsigned*)(hs + (size_t)L_ * H_);  // 2 x u32

    hipMemsetAsync(bar, 0, 2 * sizeof(unsigned), stream);

    dim3 blk(256);
    dim3 g1(NDIM / BN, MDIM / BM);   // 16 x 32 blocks

    // Phase 1: x_proj = x @ w_input   (bf16 WMMA, f32 accum)
    gemm_bf16_wmma<<<g1, blk, 0, stream>>>(x, wi, xproj);
    // Phase 2: sequential tanh recurrence (fp32, persistent kernel)
    rnn_scan<<<dim3(SCAN_BLOCKS), blk, 0, stream>>>(xproj, h0, wh, hs, bar);
    // Phase 3: outputs = hs @ w_output (bf16 WMMA, f32 accum) -> d_out
    gemm_bf16_wmma<<<g1, blk, 0, stream>>>(hs, wo, out);
    // h_final tail
    copy_hfinal<<<dim3(H_ / 256), blk, 0, stream>>>(hs, out + (size_t)L_ * H_);
}
